// xLSTMLayer_88416196756280
// MI455X (gfx1250) — matrix operand order repaired and checked by reading, not compiled, not run on hardware
//
#include <hip/hip_runtime.h>
#include <hip/hip_bf16.h>
#include <stdint.h>

// ---------------------------------------------------------------------------
// LSTM layer for MI455X (gfx1250, wave32, WMMA, async-to-LDS, NT streaming).
//   Phase 1: x_gates[TB,4H] = bf16-WMMA GEMM(input, w_ih) + (b_ih+b_hh) (fp32,
//            NT stores: written once, read once).
//   Phase 2: persistent 32-block scan. Per block: 256KB bf16 w_hh slice
//            resident in LDS; WAVE-PRIVATE async-staged h strips (16x64,
//            double-buffered) so the GEMM k-loop needs NO block barriers —
//            each wave self-syncs on its own ASYNCcnt. One fenced grid
//            barrier per step; fp32 outputs NT-stored after the barrier.
// ---------------------------------------------------------------------------

typedef __attribute__((ext_vector_type(16))) __bf16 v16bf;
typedef __attribute__((ext_vector_type(8)))  float  v8f;
typedef __attribute__((ext_vector_type(4)))  unsigned v4u;

#define T_STEPS 512
#define BATCH   128
#define DIM     1024
#define HID     1024
#define G4H     4096
#define NBLK2   32
#define HSLICE  32

union Frag16 { v16bf bf; uint4 u4[2]; };

__device__ __forceinline__ float sigmoid_f(float x) {
  return 1.0f / (1.0f + __expf(-x));
}

// ---- CDNA5 async global->LDS copy (ASYNCcnt path, ISA §15.18.3) -----------
__device__ __forceinline__ void async_copy_32B(__bf16* dstLds, const __bf16* src) {
  unsigned l0 = (unsigned)(uintptr_t)dstLds;
  asm volatile("global_load_async_to_lds_b128 %0, %1, off"
               :: "v"(l0), "v"(src) : "memory");
  unsigned l1 = l0 + 16;
  const __bf16* s1 = src + 8;
  asm volatile("global_load_async_to_lds_b128 %0, %1, off"
               :: "v"(l1), "v"(s1) : "memory");
}
__device__ __forceinline__ void wait_async0() {
  asm volatile("s_wait_asynccnt 0" ::: "memory");
}
__device__ __forceinline__ void wait_async_le4() {   // allow 1 stage (4 ops) in flight
  asm volatile("s_wait_asynccnt 4" ::: "memory");
}

// --------------------------- helper kernels --------------------------------

__global__ void f32_to_bf16_kernel(const float* __restrict__ src,
                                   __bf16* __restrict__ dst, size_t n) {
  size_t stride = (size_t)gridDim.x * blockDim.x;
  for (size_t i = (size_t)blockIdx.x * blockDim.x + threadIdx.x; i < n; i += stride)
    dst[i] = (__bf16)__builtin_nontemporal_load(src + i);
}

__global__ void prep_kernel(const float* __restrict__ b_ih,
                            const float* __restrict__ b_hh,
                            float* __restrict__ bias_sum,
                            unsigned* __restrict__ bar) {
  int i = blockIdx.x * blockDim.x + threadIdx.x;
  if (i < G4H) bias_sum[i] = b_ih[i] + b_hh[i];
  if (i == 0) bar[0] = 0u;  // count; gen (bar[1]) keeps its running value
}

// --------------------- fragment loaders (LDS, row-major) -------------------
__device__ __forceinline__ v16bf load_frag_A(const __bf16* row, int kh) {
  Frag16 f;
  f.u4[0] = *(const uint4*)(row + 16 * kh);
  f.u4[1] = *(const uint4*)(row + 16 * kh + 8);
  return f.bf;
}

// ------------------------- phase 1: x_gates GEMM ---------------------------

__global__ __launch_bounds__(256)
void gemm_xgates_kernel(const __bf16* __restrict__ A,     // [65536][1024]
                        const __bf16* __restrict__ Bnk,   // [4096][1024]
                        const float* __restrict__ bias,   // [4096]
                        float* __restrict__ Xg) {         // [65536][4096]
  __shared__ __bf16 sA[2][128 * 32];
  __shared__ __bf16 sB[2][128 * 32];
  const int nTilesN = G4H / 128;
  const int bm = blockIdx.x / nTilesN;
  const int bn = blockIdx.x % nTilesN;
  const int tid = threadIdx.x;
  const int lane = tid & 31, wave = tid >> 5;
  const int ml = lane & 15, kh = lane >> 4;
  const int sr = tid >> 1, shalf = (tid & 1) * 16;

  const __bf16* gArow = A + (size_t)(bm * 128 + sr) * DIM + shalf;
  const __bf16* gBrow = Bnk + (size_t)(bn * 128 + sr) * DIM + shalf;

  auto stage = [&](int buf, int k0) {
    async_copy_32B(&sA[buf][sr * 32 + shalf], gArow + k0);
    async_copy_32B(&sB[buf][sr * 32 + shalf], gBrow + k0);
  };

  stage(0, 0);
  wait_async0();
  __syncthreads();

  v8f acc[8] = {};
  for (int kb = 0; kb < DIM / 32; ++kb) {
    const int cur = kb & 1;
    if (kb < DIM / 32 - 1) stage(cur ^ 1, (kb + 1) * 32);

    v16bf a = load_frag_A(&sA[cur][(wave * 16 + ml) * 32], kh);
    Frag16 bf[8];
#pragma unroll
    for (int nt = 0; nt < 8; ++nt) {
      const __bf16* brow = &sB[cur][(nt * 16 + ml) * 32];
      bf[nt].u4[0] = *(const uint4*)(brow + 16 * kh);
      bf[nt].u4[1] = *(const uint4*)(brow + 16 * kh + 8);
    }
#pragma unroll
    for (int nt = 0; nt < 8; ++nt)
      acc[nt] = __builtin_amdgcn_wmma_f32_16x16x32_bf16(
          false, a, false, bf[nt].bf, (short)0, acc[nt], false, false);

    if (kb < DIM / 32 - 1) wait_async0();
    __syncthreads();
  }

  // C element (VGPR r): M = base + 8*(lane>>4) + r, N = nt*16 + (lane&15)
  const int m_base = bm * 128 + wave * 16 + kh * 8;
  const int n_lane = bn * 128 + ml;
#pragma unroll
  for (int nt = 0; nt < 8; ++nt) {
    float bs = bias[n_lane + nt * 16];
#pragma unroll
    for (int r = 0; r < 8; ++r)
      __builtin_nontemporal_store(
          acc[nt][r] + bs, &Xg[(size_t)(m_base + r) * G4H + n_lane + nt * 16]);
  }
}

// --------------------------- grid-wide barrier -----------------------------
// Release: every thread fences (drains its own stores to device scope) before
// arriving. Acquire: every thread fences after release from the barrier so
// stale L0 lines of the ping-pong h buffers are invalidated.
__device__ __forceinline__ void grid_barrier(unsigned* cnt, unsigned* gen,
                                             unsigned nblocks) {
  __threadfence();
  __syncthreads();
  if (threadIdx.x == 0) {
    unsigned g0 = __hip_atomic_load(gen, __ATOMIC_RELAXED, __HIP_MEMORY_SCOPE_AGENT);
    unsigned old = __hip_atomic_fetch_add(cnt, 1u, __ATOMIC_ACQ_REL, __HIP_MEMORY_SCOPE_AGENT);
    if (old == nblocks - 1) {
      __hip_atomic_store(cnt, 0u, __ATOMIC_RELAXED, __HIP_MEMORY_SCOPE_AGENT);
      __hip_atomic_fetch_add(gen, 1u, __ATOMIC_RELEASE, __HIP_MEMORY_SCOPE_AGENT);
    } else {
      while (__hip_atomic_load(gen, __ATOMIC_ACQUIRE, __HIP_MEMORY_SCOPE_AGENT) == g0)
        __builtin_amdgcn_s_sleep(2);
    }
  }
  __syncthreads();
  __threadfence();
}

// ------------------------- phase 2: recurrent scan -------------------------
// LDS (312 KB of 320 KB):
//   sW   bf16 w_hh slice [128][1024]        256 KB (persistent)
//   sAw  bf16 wave-private h strips
//        [8 waves][2 bufs][16][64]           32 KB (async-filled, no barriers)
//   sC   fp32 cell state [128][32]           16 KB
//   sH   bf16 h_new staging [128][32]         8 KB (coalesced pre-barrier store)

__global__ __launch_bounds__(256)
void lstm_scan_kernel(const float* __restrict__ Xg,      // [T*B][4096]
                      const __bf16* __restrict__ Whh,    // [4096][1024] bf16 [N][K]
                      const float* __restrict__ c0,
                      __bf16* __restrict__ h_ping,       // holds h0 (bf16)
                      __bf16* __restrict__ h_pong,
                      float* __restrict__ out,
                      float* __restrict__ hf,
                      float* __restrict__ cf,
                      unsigned* __restrict__ bar) {
  extern __shared__ unsigned char smem[];
  __bf16* sW  = (__bf16*)smem;                               // 262144 B
  __bf16* sAw = (__bf16*)(smem + 262144);                    //  32768 B
  float*  sC  = (float*)(smem + 262144 + 32768);             //  16384 B
  __bf16* sH  = (__bf16*)(smem + 262144 + 32768 + 16384);    //   8192 B

  const int j = blockIdx.x;
  const int tid = threadIdx.x;
  const int lane = tid & 31, wave = tid >> 5;
  const int ml = lane & 15, kh = lane >> 4;
  const int sr = tid >> 1, shalf = (tid & 1) * 16;

  __bf16* myA = sAw + wave * 2048;   // [2][16][64] elements, wave-private

  // Persistent w_hh slice: LDS row r = g*32+c <-> global row g*1024 + j*32 + c
  for (int r = wave; r < 128; r += 8) {
    int g = r >> 5, c = r & 31;
    const __bf16* src = Whh + (size_t)(g * HID + j * HSLICE + c) * HID;
    __bf16* dst = sW + r * HID;
    for (int e = lane * 8; e < HID; e += 32 * 8)
      *(uint4*)(dst + e) = *(const uint4*)(src + e);
  }
  for (int i = tid; i < 128 * 32; i += 256) {
    int m = i >> 5, c = i & 31;
    sC[i] = c0[(size_t)m * HID + j * HSLICE + c];
  }
  __syncthreads();

  const __bf16* hr = h_ping;
  __bf16* hw = h_pong;

  for (int t = 0; t < T_STEPS; ++t) {
    // Prefetch this step's x_gates slice: pulls HBM lines toward L2 so the
    // epilogue's NT loads are short-latency. 512 cachelines, 2 per thread.
    {
      int r = tid & 127, seg = (tid >> 7) * 2;
      const char* base = (const char*)(Xg + (size_t)(t * BATCH + r) * G4H + j * HSLICE);
      __builtin_prefetch(base + (size_t)seg * HID * 4, 0, 3);
      __builtin_prefetch(base + (size_t)(seg + 1) * HID * 4, 0, 3);
    }

    // Wave-private async stage of rows [wave*16, wave*16+16), 64 K at a time.
    auto stageA = [&](int buf, int k0) {
      __bf16* dst = myA + buf * 1024 + ml * 64 + kh * 16;
      const __bf16* src = hr + (size_t)(wave * 16 + ml) * HID + k0 + kh * 16;
      async_copy_32B(dst, src);
      async_copy_32B(dst + 32, src + 32);
    };

    stageA(0, 0);
    v8f acc[8] = {};
    for (int it = 0; it < HID / 64; ++it) {          // 16 iters, NO barriers
      const int cur = it & 1;
      if (it < HID / 64 - 1) {
        stageA(cur ^ 1, (it + 1) * 64);
        wait_async_le4();                            // cur tile done; next in flight
      } else {
        wait_async0();
      }
      const __bf16* arow = myA + cur * 1024 + ml * 64;
#pragma unroll
      for (int s = 0; s < 2; ++s) {
        const int kb32 = it * 64 + s * 32;
        v16bf a = load_frag_A(arow + s * 32, kh);
        Frag16 bf[8];
#pragma unroll
        for (int nt = 0; nt < 8; ++nt) {
          const __bf16* brow = sW + (nt * 16 + ml) * HID + kb32;
          bf[nt].u4[0] = *(const uint4*)(brow + 16 * kh);
          bf[nt].u4[1] = *(const uint4*)(brow + 16 * kh + 8);
        }
#pragma unroll
        for (int nt = 0; nt < 8; ++nt)
          acc[nt] = __builtin_amdgcn_wmma_f32_16x16x32_bf16(
              false, a, false, bf[nt].bf, (short)0, acc[nt], false, false);
      }
    }

    // Gate epilogue (wave-local rows): accumulator tile nt = 2*g + cp.
    const int m_base = wave * 16 + kh * 8;
    float hval[16];
#pragma unroll
    for (int cp = 0; cp < 2; ++cp) {
      const int c_local = cp * 16 + ml;
#pragma unroll
      for (int r = 0; r < 8; ++r) {
        const int m = m_base + r;
        const float* xrow =
            Xg + (size_t)(t * BATCH + m) * G4H + j * HSLICE + c_local;
        float iv = acc[0 + cp][r] + __builtin_nontemporal_load(xrow + 0 * HID);
        float fv = acc[2 + cp][r] + __builtin_nontemporal_load(xrow + 1 * HID);
        float gv = acc[4 + cp][r] + __builtin_nontemporal_load(xrow + 2 * HID);
        float ov = acc[6 + cp][r] + __builtin_nontemporal_load(xrow + 3 * HID);
        float c_old = sC[m * 32 + c_local];
        float c_new = sigmoid_f(fv) * c_old + sigmoid_f(iv) * tanhf(gv);
        float h_new = sigmoid_f(ov) * tanhf(c_new);
        sC[m * 32 + c_local] = c_new;
        sH[m * 32 + c_local] = (__bf16)h_new;
        hval[cp * 8 + r] = h_new;
      }
    }
    __syncthreads();

    // Coalesced bf16 h store — the only store that must precede the barrier.
    {
      const __bf16* s = sH + sr * 32 + shalf;
      __bf16* d = hw + (size_t)sr * HID + j * HSLICE + shalf;
      *(uint4*)d = *(const uint4*)s;
      *(uint4*)(d + 8) = *(const uint4*)(s + 8);
    }
    grid_barrier(&bar[0], &bar[1], NBLK2);

    // Deferred fp32 output stores (off the critical path, overlap next GEMM).
    {
      float* outt = out + (size_t)t * BATCH * HID;
#pragma unroll
      for (int cp = 0; cp < 2; ++cp) {
        const int c_local = cp * 16 + ml;
#pragma unroll
        for (int r = 0; r < 8; ++r) {
          size_t col = (size_t)(m_base + r) * HID + j * HSLICE + c_local;
          __builtin_nontemporal_store(hval[cp * 8 + r], outt + col);
          if (t == T_STEPS - 1) hf[col] = hval[cp * 8 + r];
        }
      }
    }

    const __bf16* tmp = hr; hr = hw; hw = (__bf16*)tmp;
  }

  // Final cell state.
  for (int i = tid; i < 128 * 32; i += 256) {
    int m = i >> 5, c = i & 31;
    cf[(size_t)m * HID + j * HSLICE + c] = sC[i];
  }
}

// ------------------------------- launcher ----------------------------------

extern "C" void kernel_launch(void* const* d_in, const int* in_sizes, int n_in,
                              void* d_out, int out_size, void* d_ws, size_t ws_size,
                              hipStream_t stream) {
  const float* input = (const float*)d_in[0];
  const float* h0    = (const float*)d_in[1];
  const float* c0    = (const float*)d_in[2];
  const float* w_ih  = (const float*)d_in[3];
  const float* w_hh  = (const float*)d_in[4];
  const float* b_ih  = (const float*)d_in[5];
  const float* b_hh  = (const float*)d_in[6];

  char* ws = (char*)d_ws;
  size_t off = 0;
  float* xg = (float*)(ws + off);        off += (size_t)T_STEPS * BATCH * G4H * 4;
  __bf16* inp_bf = (__bf16*)(ws + off);  off += (size_t)T_STEPS * BATCH * DIM * 2;
  __bf16* wih_bf = (__bf16*)(ws + off);  off += (size_t)G4H * DIM * 2;
  __bf16* whh_bf = (__bf16*)(ws + off);  off += (size_t)G4H * HID * 2;
  __bf16* h_ping = (__bf16*)(ws + off);  off += (size_t)BATCH * HID * 2;
  __bf16* h_pong = (__bf16*)(ws + off);  off += (size_t)BATCH * HID * 2;
  float* bias = (float*)(ws + off);      off += (size_t)G4H * 4;
  unsigned* bar = (unsigned*)(ws + off); off += 2 * sizeof(unsigned);

  f32_to_bf16_kernel<<<4096, 256, 0, stream>>>(input, inp_bf,
                                               (size_t)T_STEPS * BATCH * DIM);
  f32_to_bf16_kernel<<<1024, 256, 0, stream>>>(w_ih, wih_bf, (size_t)G4H * DIM);
  f32_to_bf16_kernel<<<1024, 256, 0, stream>>>(w_hh, whh_bf, (size_t)G4H * HID);
  f32_to_bf16_kernel<<<32, 256, 0, stream>>>(h0, h_ping, (size_t)BATCH * HID);
  prep_kernel<<<16, 256, 0, stream>>>(b_ih, b_hh, bias, bar);

  dim3 g1((T_STEPS * BATCH / 128) * (G4H / 128));   // 16384 blocks
  gemm_xgates_kernel<<<g1, 256, 0, stream>>>(inp_bf, wih_bf, bias, xg);

  float* out = (float*)d_out;
  float* hf = out + (size_t)T_STEPS * BATCH * HID;
  float* cf = hf + (size_t)BATCH * HID;
  lstm_scan_kernel<<<NBLK2, 256,
                     262144 + 32768 + 16384 + 8192, stream>>>(
      xg, whh_bf, c0, h_ping, h_pong, out, hf, cf, bar);
}
